// QuantLinear_59571196395668
// MI455X (gfx1250) — 8-bit matrix operand order repaired and checked by reading, compiled, not run on hardware
//
#include <hip/hip_runtime.h>
#include <hip/hip_bf16.h>
#include <stdint.h>

typedef __attribute__((ext_vector_type(16))) int   v16i;
typedef __attribute__((ext_vector_type(8)))  float v8f;
typedef __attribute__((ext_vector_type(4)))  int   v4i;
typedef __attribute__((ext_vector_type(4)))  float v4f;

#define FP8_MAX 448.0f
#define QEPS 1e-8f

#if __has_builtin(__builtin_amdgcn_cvt_pk_fp8_f32)
__device__ __forceinline__ uint32_t pack4_fp8(float a, float b, float c, float d) {
    int w = __builtin_amdgcn_cvt_pk_fp8_f32(a, b, 0, false);
    w = __builtin_amdgcn_cvt_pk_fp8_f32(c, d, w, true);
    return (uint32_t)w;
}
#else
__device__ __forceinline__ uint32_t fp8_byte(float f) {
    uint32_t u = __builtin_bit_cast(uint32_t, f);
    uint32_t sgn = (u >> 24) & 0x80u;
    float af = fabsf(f);
    int e = (int)((u >> 23) & 0xFFu) - 127;
    if (e < -6) e = -6;
    float sc = __builtin_bit_cast(float, (uint32_t)((3 - e + 127) << 23));
    int mi = (int)rintf(af * sc);
    if (mi >= 16) { e += 1; mi = 8; }
    uint32_t byte;
    if (mi < 8) byte = sgn | (uint32_t)mi;
    else        byte = sgn | (uint32_t)((e + 7) << 3) | (uint32_t)(mi - 8);
    return byte;
}
__device__ __forceinline__ uint32_t pack4_fp8(float a, float b, float c, float d) {
    return fp8_byte(a) | (fp8_byte(b) << 8) | (fp8_byte(c) << 16) | (fp8_byte(d) << 24);
}
#endif

__device__ __forceinline__ float clamp8(float v) {
    return fminf(fmaxf(v, -FP8_MAX), FP8_MAX);
}

__device__ __forceinline__ float bf16_rne(float f) {
    uint32_t u = __builtin_bit_cast(uint32_t, f);
    uint32_t r = (u + 0x7FFFu + ((u >> 16) & 1u)) & 0xFFFF0000u;
    return __builtin_bit_cast(float, r);
}

__global__ void qx_kernel(const float* __restrict__ x, uint8_t* __restrict__ qA,
                          float* __restrict__ sxT, int M, int K) {
    const int KG = K >> 7;
    int wid  = (blockIdx.x * blockDim.x + threadIdx.x) >> 5;
    int lane = threadIdx.x & 31;
    int m = wid / KG;
    int g = wid - m * KG;
    if (m >= M) return;

    const v4f v = *(const v4f*)(x + (size_t)m * K + g * 128 + lane * 4);
    float a = fmaxf(fmaxf(fabsf(v[0]), fabsf(v[1])), fmaxf(fabsf(v[2]), fabsf(v[3])));
#pragma unroll
    for (int off = 16; off >= 1; off >>= 1)
        a = fmaxf(a, __shfl_xor(a, off, 32));
    float s = fmaxf(a / FP8_MAX, QEPS);
    float inv = 1.0f / s;

    uint32_t packed = pack4_fp8(clamp8(v[0] * inv), clamp8(v[1] * inv),
                                clamp8(v[2] * inv), clamp8(v[3] * inv));

    int mp   = m & 15, tm = m >> 4;
    int half = lane >> 4;
    int j    = (lane & 15) >> 2;
    int lh   = (lane >> 2) & 1;
    int i    = lane & 1;
    int vreg = (lane >> 3) * 4 + (lane & 3);
    int lane_out = lh * 16 + mp;

    uint32_t* q32 = (uint32_t*)qA;
    q32[((size_t)(tm * KG + g)) * 512 + lane_out * 16 + vreg] = packed;
    if (lane == 0) sxT[(size_t)g * M + m] = s;
}

__global__ void qw_kernel(const float* __restrict__ W, uint8_t* __restrict__ qB,
                          float* __restrict__ sW, int N, int K) {
    const int KB = K >> 7;
    int nb = blockIdx.x / KB, kb = blockIdx.x % KB;
    int t = threadIdx.x;
    __shared__ float red[256];
    __shared__ float sv;

    const float* base = W + (size_t)nb * 128 * K + kb * 128;
    float a = 0.0f;
#pragma unroll
    for (int it = 0; it < 16; ++it) {
        int f = t + 256 * it;
        int np = f >> 5, k4 = f & 31;
        v4f v = *(const v4f*)(base + (size_t)np * K + k4 * 4);
        a = fmaxf(a, fmaxf(fmaxf(fabsf(v[0]), fabsf(v[1])),
                           fmaxf(fabsf(v[2]), fabsf(v[3]))));
    }
    red[t] = a;
    __syncthreads();
    for (int off = 128; off >= 1; off >>= 1) {
        if (t < off) red[t] = fmaxf(red[t], red[t + off]);
        __syncthreads();
    }
    if (t == 0) {
        float s = fmaxf(red[0] / FP8_MAX, QEPS);
        sv = s;
        sW[nb * KB + kb] = s;
    }
    __syncthreads();
    float inv = 1.0f / sv;

    uint32_t* q32 = (uint32_t*)qB;
#pragma unroll
    for (int it = 0; it < 16; ++it) {
        int f = t + 256 * it;
        int np = f >> 5, k4 = f & 31;
        v4f v = *(const v4f*)(base + (size_t)np * K + k4 * 4);
        uint32_t packed = pack4_fp8(clamp8(v[0] * inv), clamp8(v[1] * inv),
                                    clamp8(v[2] * inv), clamp8(v[3] * inv));
        int nn = np & 15;
        int tn = nb * 8 + (np >> 4);
        int G  = k4 >> 3;
        int lh = (k4 >> 2) & 1;
        int d  = k4 & 3;
        q32[((size_t)(tn * KB + kb)) * 512 + (lh * 16 + nn) * 16 + (G * 4 + d)] = packed;
    }
}

#define LDS_TILE   2560
#define LDS_BUF    (24 * LDS_TILE)
#define LDS_TOTAL  (2 * LDS_BUF)

extern __shared__ uint8_t smem[];

__device__ __forceinline__ v16i lds_frag(const uint8_t* p) {
    v16i r;
    v4i a = *(const v4i*)(p);
    v4i b = *(const v4i*)(p + 16);
    v4i c = *(const v4i*)(p + 32);
    v4i d = *(const v4i*)(p + 48);
#pragma unroll
    for (int i = 0; i < 4; ++i) {
        r[i] = a[i]; r[4 + i] = b[i]; r[8 + i] = c[i]; r[12 + i] = d[i];
    }
    return r;
}

__global__ void __launch_bounds__(256)
gemm_kernel(const uint8_t* __restrict__ qA, const uint8_t* __restrict__ qB,
            const float* __restrict__ sxT, const float* __restrict__ sW,
            const float* __restrict__ bias, float* __restrict__ out,
            int M, int N, int K) {
    const int KG = K >> 7;
    int t = threadIdx.x;
    int wave = t >> 5, lane = t & 31;
    int lh = lane >> 4, ln = lane & 15;
    int nblk = N >> 8;
    int brow = blockIdx.x / nblk;
    int bcol = blockIdx.x % nblk;
    int wr = wave >> 2, wc = wave & 3;
    int m0 = brow * 128 + wr * 64;
    int n0 = bcol * 256 + wc * 64;
    int tm0 = brow * 8;
    int tn0 = bcol * 16;

    uint32_t lds_base = (uint32_t)(uintptr_t)smem;

    auto stage = [&](int g, int buf) {
        uint32_t bufb = lds_base + buf * LDS_BUF;
#pragma unroll
        for (int c = 0; c < 12; ++c) {
            int flat = t + 256 * c;
            int tile = flat >> 7;
            int q    = flat & 127;
            uint32_t loff = bufb + tile * LDS_TILE + (q >> 2) * 80 + (q & 3) * 16;
            const uint8_t* src = (tile < 8)
                ? qA + ((((size_t)(tm0 + tile)) * KG + g) << 11) + q * 16
                : qB + ((((size_t)(tn0 + tile - 8)) * KG + g) << 11) + q * 16;
            asm volatile("global_load_async_to_lds_b128 %0, %1, off"
                         :: "v"(loff), "v"(src) : "memory");
        }
    };

    v8f acc[4][4];
#pragma unroll
    for (int mt = 0; mt < 4; ++mt)
#pragma unroll
        for (int nt = 0; nt < 4; ++nt)
            acc[mt][nt] = (v8f){0.f, 0.f, 0.f, 0.f, 0.f, 0.f, 0.f, 0.f};
    const v8f czero = (v8f){0.f, 0.f, 0.f, 0.f, 0.f, 0.f, 0.f, 0.f};

    stage(0, 0);

    for (int g = 0; g < KG; ++g) {
        int buf = g & 1;
        asm volatile("s_wait_asynccnt 0x0" ::: "memory");
        __syncthreads();
        if (g + 1 < KG) stage(g + 1, buf ^ 1);

        float swg = sW[(n0 >> 7) * KG + g];

        float sc[4][8];
#pragma unroll
        for (int mt = 0; mt < 4; ++mt) {
            const v4f* sp = (const v4f*)(sxT + (size_t)g * M + m0 + mt * 16 + lh * 8);
            v4f s0 = sp[0], s1 = sp[1];
#pragma unroll
            for (int r = 0; r < 4; ++r) {
                sc[mt][r]     = s0[r] * swg;
                sc[mt][4 + r] = s1[r] * swg;
            }
        }

        const uint8_t* bufp = smem + buf * LDS_BUF;
        v16i bf[4];
#pragma unroll
        for (int nt = 0; nt < 4; ++nt)
            bf[nt] = lds_frag(bufp + (8 + wc * 4 + nt) * LDS_TILE + lane * 80);

#pragma unroll
        for (int mt = 0; mt < 4; ++mt) {
            v16i af = lds_frag(bufp + (wr * 4 + mt) * LDS_TILE + lane * 80);
#pragma unroll
            for (int nt = 0; nt < 4; ++nt) {
                v8f p = __builtin_amdgcn_wmma_f32_16x16x128_fp8_fp8(
                    af, bf[nt], (short)0, czero, false, false);
#pragma unroll
                for (int r = 0; r < 8; ++r)
                    acc[mt][nt][r] += sc[mt][r] * p[r];
            }
        }
    }

#pragma unroll
    for (int nt = 0; nt < 4; ++nt) {
        int n = n0 + nt * 16 + ln;
        float bv = bias[n];
#pragma unroll
        for (int mt = 0; mt < 4; ++mt) {
#pragma unroll
            for (int r = 0; r < 8; ++r) {
                int m = m0 + mt * 16 + lh * 8 + r;
                out[(size_t)m * N + n] = bf16_rne(acc[mt][nt][r] + bv);
            }
        }
    }
}

extern "C" void kernel_launch(void* const* d_in, const int* in_sizes, int n_in,
                              void* d_out, int out_size, void* d_ws, size_t ws_size,
                              hipStream_t stream) {
    const float* x    = (const float*)d_in[0];
    const float* W    = (const float*)d_in[1];
    const float* bias = (const float*)d_in[2];
    float* out = (float*)d_out;

    const int K = 4096, N = 4096;
    const int M = in_sizes[0] / K;
    const int KG = K / 128;

    uint8_t* ws  = (uint8_t*)d_ws;
    uint8_t* qA  = ws;
    uint8_t* qB  = ws + (size_t)M * K;
    float*   sxT = (float*)(ws + (size_t)M * K + (size_t)N * K);
    float*   sWp = sxT + (size_t)KG * M;

    qx_kernel<<<(M * KG) / 8, 256, 0, stream>>>(x, qA, sxT, M, K);
    qw_kernel<<<(N / 128) * KG, 256, 0, stream>>>(W, qB, sWp, N, K);
    gemm_kernel<<<(M / 128) * (N / 256), 256, LDS_TOTAL, stream>>>(
        qA, qB, sxT, sWp, bias, out, M, N, K);
}
